// ShifthWiseConv2d_LF_60035052863997
// MI455X (gfx1250) — matrix operand order repaired and checked by reading, not compiled, not run on hardware
//
#include <hip/hip_runtime.h>
#include <hip/hip_bf16.h>

typedef __attribute__((ext_vector_type(16))) _Float16 v16h;
typedef __attribute__((ext_vector_type(8)))  _Float16 v8h;
typedef __attribute__((ext_vector_type(8)))  float    v8f;

constexpr int kCin  = 960;
constexpr int kCout = 96;
constexpr int kNK   = 5;
constexpr int kW    = 56;
constexpr int kSpat = 56 * 56;   // 3136
constexpr int kB    = 16;
constexpr int kN    = kB * kSpat; // 50176
constexpr int kArow = 40;        // padded LDS row stride (halfs) for A slice

// ---------------- kernel 1: proj_w f32 -> f16 (A matrix, 96x960 row-major) --
__global__ void wconv_kernel(const float* __restrict__ W, _Float16* __restrict__ A) {
    int i = blockIdx.x * 256 + threadIdx.x;
    if (i < kCout * kCin) A[i] = (_Float16)W[i];
}

// ---------------- kernel 2: build Bmat[n][c] (f16, 50176 x 960) -------------
// local = antialiased-bilinear resize (58->56, after replicate pad) of
//         x*(scale+1e-5)+bias, transposed to K-contiguous layout via LDS.
__global__ void stage_kernel(const float* __restrict__ x,
                             const float* __restrict__ ls,
                             const float* __restrict__ lb,
                             _Float16* __restrict__ Bm) {
    __shared__ float lds[16 * 257];
    const int tid   = threadIdx.x;
    const int n     = blockIdx.x * 256 + tid;
    const int cbase = blockIdx.y * 16;
    const int b  = n / kSpat;
    const int hw = n % kSpat;
    const int i  = hw / kW;
    const int j  = hw % kW;

    const float S = 56.0f / 58.0f;   // antialias triangle rate (downscale)

    // vertical taps (padded coords 0..57 -> x rows via clamp(k-1,0,55))
    float sy = (i + 0.5f) * (58.0f / 56.0f) - 0.5f;
    int   kby = (int)floorf(sy + 0.5f);
    int ry[3]; float wy[3]; float sumy = 0.0f;
#pragma unroll
    for (int t = 0; t < 3; ++t) {
        int k = kby - 1 + t;
        float w = fmaxf(1.0f - fabsf(sy - (float)k) * S, 0.0f);
        if (k < 0 || k > 57) w = 0.0f;
        int r = k - 1; r = r < 0 ? 0 : (r > 55 ? 55 : r);
        ry[t] = r; wy[t] = w; sumy += w;
    }
#pragma unroll
    for (int t = 0; t < 3; ++t) wy[t] /= sumy;

    // horizontal taps
    float sx = (j + 0.5f) * (58.0f / 56.0f) - 0.5f;
    int   kbx = (int)floorf(sx + 0.5f);
    int rx[3]; float wx[3]; float sumx = 0.0f;
#pragma unroll
    for (int t = 0; t < 3; ++t) {
        int k = kbx - 1 + t;
        float w = fmaxf(1.0f - fabsf(sx - (float)k) * S, 0.0f);
        if (k < 0 || k > 57) w = 0.0f;
        int r = k - 1; r = r < 0 ? 0 : (r > 55 ? 55 : r);
        rx[t] = r; wx[t] = w; sumx += w;
    }
#pragma unroll
    for (int t = 0; t < 3; ++t) wx[t] /= sumx;

    // phase 1: coalesced reads, LDS tile [c][n]
    for (int cc = 0; cc < 16; ++cc) {
        const int c = cbase + cc;
        const float* xc = x + ((size_t)b * kCin + c) * kSpat;
        float v = 0.0f;
#pragma unroll
        for (int ty = 0; ty < 3; ++ty) {
            const float* row = xc + ry[ty] * kW;
            float hv = 0.0f;
#pragma unroll
            for (int tx = 0; tx < 3; ++tx) hv += wx[tx] * row[rx[tx]];
            v += wy[ty] * hv;
        }
        lds[cc * 257 + tid] = v * (ls[c] + 1e-5f) + lb[c];
    }
    __syncthreads();

    // phase 2: transpose out of LDS, 16B K-contiguous f16 writes
#pragma unroll
    for (int it = 0; it < 2; ++it) {
        const int id = tid + it * 256;
        const int nl = id >> 1;
        const int ch = id & 1;
        v8h pk;
#pragma unroll
        for (int u = 0; u < 8; ++u) pk[u] = (_Float16)lds[(ch * 8 + u) * 257 + nl];
        const int gn = blockIdx.x * 256 + nl;
        *(v8h*)(Bm + (size_t)gn * kCin + cbase + ch * 8) = pk;
    }
}

// ---------------- kernel 3: shift/global path, writes fa/3 * sum ------------
__global__ void shift_kernel(const float* __restrict__ x,
                             const int* __restrict__ padhv,   // (480,4): [slot][g],[slot][2+g]
                             const int* __restrict__ idi,     // (480,2)
                             const float* __restrict__ fap,
                             float* __restrict__ out) {
    int idx = blockIdx.x * 256 + threadIdx.x;
    if (idx >= kB * kCout * kSpat) return;
    const int w  = idx % kW;
    const int h  = (idx / kW) % kW;
    const int co = (idx / kSpat) % kCout;
    const int b  = idx / (kSpat * kCout);
    const float* xb = x + (size_t)b * kCin * kSpat;

    float s = 0.0f;
#pragma unroll
    for (int g = 0; g < 2; ++g) {
        for (int k = 0; k < kNK; ++k) {
            const int slot = co * kNK + k;
            const float* xc = xb + (size_t)(g * 480 + slot) * kSpat;
            const int sh = padhv[slot * 4 + g];
            const int sv = padhv[slot * 4 + 2 + g];
            const int wsh = w + sh;
            if (wsh >= -1 && wsh <= 56) {
                int ww = wsh < 0 ? 0 : (wsh > 55 ? 55 : wsh);
                s += xc[h * kW + ww];
            }
            const int hsv = h + sv;
            if (hsv >= -1 && hsv <= 56) {
                int hh = hsv < 0 ? 0 : (hsv > 55 ? 55 : hsv);
                s += xc[hh * kW + w];
            }
            if (idi[slot * 2 + g] >= 0) s += xc[h * kW + w];
        }
    }
    out[idx] = (*fap) * (1.0f / 3.0f) * s;
}

// ---------------- kernel 4: WMMA projection GEMM + fused epilogue -----------
// out[o,n] += (1-fa) * sum_c A[o,c] * Bmat[n,c].
// Block = 8 waves; each wave owns a 96(M) x 32(N) strip. The shared A K-slice
// (96x32 f16) is double-buffered in LDS, loaded once per block per K step.
// All 6 A fragments are hoisted into registers so the 12 ds_loads clause
// together and the 12 WMMAs issue back-to-back.
__global__ void gemm_kernel(const _Float16* __restrict__ A,
                            const _Float16* __restrict__ Bm,
                            float* __restrict__ out,
                            const float* __restrict__ fap) {
    __shared__ _Float16 As[2][kCout * kArow];   // 2 x 7680 B

    const int tid   = threadIdx.x;
    const int lane  = tid & 31;
    const int wv    = tid >> 5;                 // wave in block, 0..7
    const int nbase = blockIdx.x * 256 + wv * 32;
    const int l16   = lane & 15;
    const int hsel  = lane >> 4;

    // B frag: lane n=l16, K = hsel*16 + {0..15}
    const _Float16* b0 = Bm + (size_t)(nbase + l16) * kCin + hsel * 16;
    const _Float16* b1 = b0 + (size_t)16 * kCin;

    // cooperative A-slice loader: 384 16B chunks; chunk c -> row c>>2, seg c&3
    const int row0 = tid >> 2, seg0 = tid & 3;
    const int c1   = 256 + tid;
    const int row1 = c1 >> 2,  seg1 = c1 & 3;

    auto load_slice = [&](int buf, int kk) {
        *(v8h*)(&As[buf][row0 * kArow + seg0 * 8]) =
            *(const v8h*)(A + (size_t)row0 * kCin + kk + seg0 * 8);
        if (tid < 128)
            *(v8h*)(&As[buf][row1 * kArow + seg1 * 8]) =
                *(const v8h*)(A + (size_t)row1 * kCin + kk + seg1 * 8);
    };

    v8f acc[12];
    const v8f zero = {0, 0, 0, 0, 0, 0, 0, 0};
#pragma unroll
    for (int q = 0; q < 12; ++q) acc[q] = zero;

    load_slice(0, 0);
    __syncthreads();

#pragma unroll 1
    for (int it = 0; it < kCin / 32; ++it) {
        const int kk  = it * 32;
        const int buf = it & 1;
        if (kk + 32 < kCin) load_slice(buf ^ 1, kk + 32);

        // B fragments (global, streaming)
        v16h bf0, bf1;
        *(v8h*)&bf0         = *(const v8h*)(b0 + kk);
        *(((v8h*)&bf0) + 1) = *(const v8h*)(b0 + kk + 8);
        *(v8h*)&bf1         = *(const v8h*)(b1 + kk);
        *(((v8h*)&bf1) + 1) = *(const v8h*)(b1 + kk + 8);

        // hoist all 6 A fragments (LDS) into distinct registers
        v16h af[6];
#pragma unroll
        for (int mt = 0; mt < 6; ++mt) {
            // A frag: lane m=l16, K = hsel*16 + {0..15}
            const _Float16* ap = &As[buf][(mt * 16 + l16) * kArow + hsel * 16];
            *(v8h*)&af[mt]         = *(const v8h*)(ap);
            *(((v8h*)&af[mt]) + 1) = *(const v8h*)(ap + 8);
        }

#pragma unroll
        for (int mt = 0; mt < 6; ++mt) {
            acc[mt * 2 + 0] = __builtin_amdgcn_wmma_f32_16x16x32_f16(
                false, af[mt], false, bf0, (short)0, acc[mt * 2 + 0], false, false);
            acc[mt * 2 + 1] = __builtin_amdgcn_wmma_f32_16x16x32_f16(
                false, af[mt], false, bf1, (short)0, acc[mt * 2 + 1], false, false);
        }
        __syncthreads();   // stores to buf^1 visible; all reads of buf done
    }

    const float beta = 1.0f - *fap;
#pragma unroll
    for (int nt = 0; nt < 2; ++nt) {
        const int n  = nbase + nt * 16 + l16;
        const int b  = n / kSpat;
        const int hw = n % kSpat;
        float* ob = out + (size_t)b * kCout * kSpat + hw;
#pragma unroll
        for (int mt = 0; mt < 6; ++mt) {
#pragma unroll
            for (int r = 0; r < 8; ++r) {
                const int m = mt * 16 + hsel * 8 + r;
                ob[(size_t)m * kSpat] += beta * acc[mt * 2 + nt][r];
            }
        }
    }
}

extern "C" void kernel_launch(void* const* d_in, const int* in_sizes, int n_in,
                              void* d_out, int out_size, void* d_ws, size_t ws_size,
                              hipStream_t stream) {
    const float* x     = (const float*)d_in[0];
    const int*   padhv = (const int*)d_in[1];
    const int*   idi   = (const int*)d_in[2];
    const float* ls    = (const float*)d_in[3];
    const float* lb    = (const float*)d_in[4];
    const float* pw    = (const float*)d_in[5];
    const float* fap   = (const float*)d_in[6];
    float* out = (float*)d_out;

    _Float16* Amat = (_Float16*)d_ws;                                     // 180 KB
    _Float16* Bmat = (_Float16*)((char*)d_ws + (size_t)kCout * kCin * 2); // 96 MB

    wconv_kernel<<<(kCout * kCin + 255) / 256, 256, 0, stream>>>(pw, Amat);

    dim3 sg(kN / 256, kCin / 16);
    stage_kernel<<<sg, 256, 0, stream>>>(x, ls, lb, Bmat);

    shift_kernel<<<(kB * kCout * kSpat + 255) / 256, 256, 0, stream>>>(x, padhv, idi, fap, out);

    gemm_kernel<<<kN / 256, 256, 0, stream>>>(Amat, Bmat, out, fap);
}
